// SlotAttention_57552561767045
// MI455X (gfx1250) — matrix operand order repaired and checked by reading, not compiled, not run on hardware
//
#include <hip/hip_runtime.h>

#define Bn 32
#define Nn 4096
#define Dn 256
#define Hn 256
#define Sn 11
#define Mn 512

typedef __bf16 bf16_t;
typedef __attribute__((ext_vector_type(16))) __bf16 v16bf;
typedef __attribute__((ext_vector_type(8)))  float  v8f;
typedef __attribute__((ext_vector_type(4)))  unsigned int v4u;
typedef __attribute__((ext_vector_type(8)))  int v8i_t;
typedef __attribute__((ext_vector_type(4)))  int v4i_t;

union Frag32B { v16bf v; uint4 q[2]; };

__device__ inline v8f zero_v8f() {
  v8f z = {0.f,0.f,0.f,0.f,0.f,0.f,0.f,0.f};
  return z;
}

// A fragment (16x32 bf16, M x K), row-major source, 16B-aligned rows.
// lanes 0-15: row=lane, v0-3 K=0..7, v4-7 K=16..23 ; lanes 16-31: K=8..15 / 24..31
__device__ inline v16bf load_frag_a_bf16(const bf16_t* tile, int ld) {
  const int lane = threadIdx.x & 31;
  const int row  = lane & 15;
  const int hf   = lane >> 4;
  const bf16_t* p = tile + row * ld + hf * 8;
  Frag32B f;
  f.q[0] = *reinterpret_cast<const uint4*>(p);
  f.q[1] = *reinterpret_cast<const uint4*>(p + 16);
  return f.v;
}

// A fragment built from f32 source (on-the-fly bf16 convert)
__device__ inline v16bf load_frag_a_f32(const float* tile, int ld) {
  const int lane = threadIdx.x & 31;
  const int row  = lane & 15;
  const int hf   = lane >> 4;
  const float* p = tile + row * ld + hf * 8;
  Frag32B f;
#pragma unroll
  for (int i = 0; i < 8; ++i) f.v[i] = (bf16_t)p[i];
#pragma unroll
  for (int i = 0; i < 8; ++i) f.v[8 + i] = (bf16_t)p[16 + i];
  return f.v;
}

// B fragment (32x16 bf16, K x N) from K-contiguous ("Bt" row-major [N][K]) source.
// lanes 0-15: col=lane, K=0..7 / 16..23 ; lanes 16-31: K=8..15 / 24..31
__device__ inline v16bf load_frag_b_bf16(const bf16_t* tile, int ld) {
  const int lane = threadIdx.x & 31;
  const int col  = lane & 15;
  const int hf   = lane >> 4;
  const bf16_t* p = tile + col * ld + hf * 8;
  Frag32B f;
  f.q[0] = *reinterpret_cast<const uint4*>(p);
  f.q[1] = *reinterpret_cast<const uint4*>(p + 16);
  return f.v;
}

__device__ inline v8f wmma_bf16(v16bf a, v16bf b, v8f c) {
  return __builtin_amdgcn_wmma_f32_16x16x32_bf16(false, a, false, b, (short)0, c,
                                                 false, false);
}

__device__ inline float sigm(float x) { return 1.f / (1.f + __expf(-x)); }

// ---------------------------------------------------------------------------
// Tensor Data Mover: 2D tile (bf16) from global -> LDS.  D# built per ISA ch.8:
//  group0: [1:0]=count=1, [63:32]=lds_addr, [120:64]=global_addr, [127:126]=2
//  group1: [17:16]=data_size(1->2B), [79:48]=tensor_dim0, [111:80]=tensor_dim1,
//          [127:112]=tile_dim0, [143:128]=tile_dim1, [207:160]=dim0_stride
// Dims/strides in data_size units.  workgroup_mask=0 (not in a cluster).
// ---------------------------------------------------------------------------
__device__ inline void tdm_load_2d_bf16(unsigned int lds_off, const bf16_t* gaddr,
                                        unsigned int tensor_d0, unsigned int tensor_d1,
                                        unsigned int tile_d0, unsigned int tile_d1,
                                        unsigned int stride_d0) {
  unsigned long long ga = (unsigned long long)(uintptr_t)gaddr;
  v4u g0;
  g0[0] = 1u;                                                   // count=1
  g0[1] = lds_off;                                              // lds_addr (bytes)
  g0[2] = (unsigned int)ga;                                     // global_addr[31:0]
  g0[3] = (unsigned int)((ga >> 32) & 0x01FFFFFFu) | (2u << 30);// [56:32] | type=2
  v8i_t g1;
  g1[0] = (int)(1u << 16);                                      // data_size = 2B
  g1[1] = (int)(tensor_d0 << 16);                               // tensor_dim0[15:0]
  g1[2] = (int)((tensor_d0 >> 16) | (tensor_d1 << 16));         // d0[31:16] | d1[15:0]
  g1[3] = (int)((tensor_d1 >> 16) | (tile_d0 << 16));           // d1[31:16] | tile0
  g1[4] = (int)(tile_d1 & 0xFFFFu);                             // tile1 | tile2=0
  g1[5] = (int)stride_d0;                                       // dim0_stride[31:0]
  g1[6] = 0;                                                    // stride hi | d1_stride
  g1[7] = 0;
  v4i_t z4 = {0, 0, 0, 0};
#if __clang_major__ >= 23
  v8i_t z8 = {0, 0, 0, 0, 0, 0, 0, 0};
  __builtin_amdgcn_tensor_load_to_lds(g0, g1, z4, z4, z8, 0);
#else
  __builtin_amdgcn_tensor_load_to_lds(g0, g1, z4, z4, 0);
#endif
}

// ---------------- weight transpose + bf16 pack:  out[c][r] = in[r][c] ----------
__global__ void k_transpose_bf16(const float* __restrict__ in, bf16_t* __restrict__ out,
                                 int R, int C) {
  int i = blockIdx.x * 256 + threadIdx.x;
  if (i < R * C) {
    int r = i / C, c = i - r * C;
    out[(size_t)c * R + r] = (bf16_t)in[i];
  }
}

// ---------------- slots = mu + exp(log_sigma) * noise --------------------------
__global__ void k_init_slots(const float* __restrict__ noise, const float* __restrict__ mu,
                             const float* __restrict__ ls, float* __restrict__ slots) {
  int i = blockIdx.x * 256 + threadIdx.x;  // 352*256 total
  int h = i & (Hn - 1);
  slots[i] = mu[h] + __expf(ls[h]) * noise[i];
}

// ---------------- fused input LN + kf/vf projection ----------------------------
// grid (Nn/16, Bn), block 256 (8 waves). Each block: 16 rows of x.
__global__ void __launch_bounds__(256) k_ln_project(
    const float* __restrict__ inp, const float* __restrict__ lnsc,
    const float* __restrict__ lnof, const bf16_t* __restrict__ wk_t,
    const bf16_t* __restrict__ wv_t, bf16_t* __restrict__ kf,
    bf16_t* __restrict__ vf_t) {
  __shared__ __align__(16) bf16_t xs[16 * Dn];   // normalized rows (bf16)  8KB
  __shared__ float red0[16][16];
  __shared__ float red1[16][16];

  const int b   = blockIdx.y;
  const int n0  = blockIdx.x * 16;
  const int tid = threadIdx.x;
  const int row = tid >> 4;   // 0..15
  const int seg = tid & 15;   // 16 elements each

  const float* src = inp + ((size_t)b * Nn + n0 + row) * Dn + seg * 16;
  float x[16];
  float s = 0.f, ss = 0.f;
#pragma unroll
  for (int i = 0; i < 16; ++i) { x[i] = src[i]; s += x[i]; ss += x[i] * x[i]; }
  red0[row][seg] = s;
  red1[row][seg] = ss;
  __syncthreads();
  float mean = 0.f, m2 = 0.f;
#pragma unroll
  for (int i = 0; i < 16; ++i) { mean += red0[row][i]; m2 += red1[row][i]; }
  mean *= (1.f / Dn);
  m2 = m2 * (1.f / Dn) - mean * mean;
  const float inv = rsqrtf(m2 + 1e-5f);
#pragma unroll
  for (int i = 0; i < 16; ++i) {
    int c = seg * 16 + i;
    xs[row * Dn + c] = (bf16_t)((x[i] - mean) * lnsc[c] * inv + lnof[c]);
  }
  __syncthreads();

  // 8 waves x 4 tiles: tiles 0..15 -> kf h-tiles, 16..31 -> vf h-tiles
  const int wid = tid >> 5, lane = tid & 31;
  v8f acc0 = zero_v8f(), acc1 = zero_v8f(), acc2 = zero_v8f(), acc3 = zero_v8f();
  for (int k0 = 0; k0 < Dn; k0 += 32) {
    v16bf a = load_frag_a_bf16(xs + k0, Dn);
    {
      int t = wid * 4 + 0; const bf16_t* wt = (t < 16) ? wk_t : wv_t; int ht = t & 15;
      acc0 = wmma_bf16(a, load_frag_b_bf16(wt + (size_t)ht * 16 * Dn + k0, Dn), acc0);
    }
    {
      int t = wid * 4 + 1; const bf16_t* wt = (t < 16) ? wk_t : wv_t; int ht = t & 15;
      acc1 = wmma_bf16(a, load_frag_b_bf16(wt + (size_t)ht * 16 * Dn + k0, Dn), acc1);
    }
    {
      int t = wid * 4 + 2; const bf16_t* wt = (t < 16) ? wk_t : wv_t; int ht = t & 15;
      acc2 = wmma_bf16(a, load_frag_b_bf16(wt + (size_t)ht * 16 * Dn + k0, Dn), acc2);
    }
    {
      int t = wid * 4 + 3; const bf16_t* wt = (t < 16) ? wk_t : wv_t; int ht = t & 15;
      acc3 = wmma_bf16(a, load_frag_b_bf16(wt + (size_t)ht * 16 * Dn + k0, Dn), acc3);
    }
  }
  const int col = lane & 15, hf = lane >> 4;
  v8f accs[4] = {acc0, acc1, acc2, acc3};
#pragma unroll
  for (int j = 0; j < 4; ++j) {
    int t = wid * 4 + j;
    int ht = t & 15;
#pragma unroll
    for (int i = 0; i < 8; ++i) {
      int m = i + hf * 8;
      float v = accs[j][i];
      if (t < 16)
        kf[(size_t)b * Nn * Hn + (size_t)(n0 + m) * Hn + ht * 16 + col] = (bf16_t)v;
      else
        vf_t[(size_t)b * Hn * Nn + (size_t)(ht * 16 + col) * Nn + (n0 + m)] = (bf16_t)v;
    }
  }
}

// ---------------- generic row LN over [R][256] ---------------------------------
__global__ void __launch_bounds__(256) k_ln_rows(const float* __restrict__ in,
                                                 const float* __restrict__ sc,
                                                 const float* __restrict__ of,
                                                 float* __restrict__ out) {
  __shared__ float red[256];
  const int r = blockIdx.x, t = threadIdx.x;
  const float x = in[(size_t)r * Hn + t];
  red[t] = x;
  __syncthreads();
  for (int st = 128; st > 0; st >>= 1) {
    if (t < st) red[t] += red[t + st];
    __syncthreads();
  }
  const float mean = red[0] * (1.f / Hn);
  __syncthreads();
  red[t] = x * x;
  __syncthreads();
  for (int st = 128; st > 0; st >>= 1) {
    if (t < st) red[t] += red[t + st];
    __syncthreads();
  }
  const float var = red[0] * (1.f / Hn) - mean * mean;
  out[(size_t)r * Hn + t] = (x - mean) * sc[t] * rsqrtf(var + 1e-5f) + of[t];
}

// ---------------- generic WMMA GEMM: C = A(f32 [Mr][K]) * Bt(bf16 [Nc][K]) -----
__global__ void __launch_bounds__(256) k_gemm(const float* __restrict__ A,
                                              const bf16_t* __restrict__ Bt,
                                              const float* __restrict__ bias,
                                              const float* __restrict__ resid,
                                              float* __restrict__ Cout, int Mr, int K,
                                              int Nc, int relu) {
  const int nt = Nc >> 4;
  const int gid = blockIdx.x * 8 + (threadIdx.x >> 5);
  const int mtile = gid / nt, ntile = gid - mtile * nt;
  if (mtile * 16 >= Mr) return;
  const float*  At = A + (size_t)mtile * 16 * K;
  const bf16_t* Bb = Bt + (size_t)ntile * 16 * K;
  v8f acc = zero_v8f();
  for (int k0 = 0; k0 < K; k0 += 32)
    acc = wmma_bf16(load_frag_a_f32(At + k0, K), load_frag_b_bf16(Bb + k0, K), acc);
  const int lane = threadIdx.x & 31, col = lane & 15, hf = lane >> 4;
  const int c = ntile * 16 + col;
  const float bv = bias ? bias[c] : 0.f;
#pragma unroll
  for (int i = 0; i < 8; ++i) {
    int m = mtile * 16 + i + hf * 8;
    float v = acc[i] + bv;
    if (resid) v += resid[(size_t)m * Nc + c];
    if (relu) v = fmaxf(v, 0.f);
    Cout[(size_t)m * Nc + c] = v;
  }
}

// ---------------- q = LN(slots) @ Wq * H^-0.5 -> bf16 [b][16(pad)][H] ----------
__global__ void __launch_bounds__(256) k_qproj(const float* __restrict__ A,
                                               const bf16_t* __restrict__ wq_t,
                                               bf16_t* __restrict__ qbuf) {
  const int gid = blockIdx.x * 8 + (threadIdx.x >> 5);  // 352 tiles: 22m x 16n
  const int mtile = gid >> 4, ntile = gid & 15;
  const float*  At = A + (size_t)mtile * 16 * Hn;
  const bf16_t* Bb = wq_t + (size_t)ntile * 16 * Hn;
  v8f acc = zero_v8f();
  for (int k0 = 0; k0 < Hn; k0 += 32)
    acc = wmma_bf16(load_frag_a_f32(At + k0, Hn), load_frag_b_bf16(Bb + k0, Hn), acc);
  const int lane = threadIdx.x & 31, col = lane & 15, hf = lane >> 4;
#pragma unroll
  for (int i = 0; i < 8; ++i) {
    int r = mtile * 16 + i + hf * 8;  // r in [0,352)
    int b = r / Sn, s = r - b * Sn;
    qbuf[((size_t)b * 16 + s) * Hn + ntile * 16 + col] = (bf16_t)(acc[i] * 0.0625f);
  }
}

// ---------------- logits[b][s][n] = sum_h kf[b,n,h] * q[b,s,h] -----------------
__global__ void __launch_bounds__(256) k_logits(const bf16_t* __restrict__ kf,
                                                const bf16_t* __restrict__ qbuf,
                                                float* __restrict__ logits) {
  const int b = blockIdx.y;
  const int mtile = blockIdx.x * 8 + (threadIdx.x >> 5);  // n-tile 0..255
  const bf16_t* At = kf + (size_t)b * Nn * Hn + (size_t)mtile * 16 * Hn;
  const bf16_t* Bb = qbuf + (size_t)b * 16 * Hn;
  v8f acc = zero_v8f();
  for (int k0 = 0; k0 < Hn; k0 += 32)
    acc = wmma_bf16(load_frag_a_bf16(At + k0, Hn), load_frag_b_bf16(Bb + k0, Hn), acc);
  const int lane = threadIdx.x & 31, s = lane & 15, hf = lane >> 4;
  if (s < Sn) {
    float* dst = logits + ((size_t)b * Sn + s) * Nn + mtile * 16 + hf * 8;
    *reinterpret_cast<float4*>(dst)     = make_float4(acc[0], acc[1], acc[2], acc[3]);
    *reinterpret_cast<float4*>(dst + 4) = make_float4(acc[4], acc[5], acc[6], acc[7]);
  }
}

// ---------------- softmax over s (axis=-1), +eps, in place ---------------------
__global__ void k_softmax_s(float* __restrict__ attn) {
  const int b = blockIdx.y;
  const int n = blockIdx.x * 256 + threadIdx.x;
  float* base = attn + (size_t)b * Sn * Nn + n;
  float v[Sn], mx = -3.0e38f;
#pragma unroll
  for (int s = 0; s < Sn; ++s) { v[s] = base[(size_t)s * Nn]; mx = fmaxf(mx, v[s]); }
  float sum = 0.f;
#pragma unroll
  for (int s = 0; s < Sn; ++s) { v[s] = __expf(v[s] - mx); sum += v[s]; }
  const float inv = 1.f / sum;
#pragma unroll
  for (int s = 0; s < Sn; ++s) base[(size_t)s * Nn] = v[s] * inv + 1e-8f;
}

// ---------------- sums[b][s] = sum_n attn[b][s][n] -----------------------------
__global__ void __launch_bounds__(256) k_rowsum(const float* __restrict__ attn,
                                                float* __restrict__ sums) {
  const int b = blockIdx.y, s = blockIdx.x;
  const float* row = attn + ((size_t)b * Sn + s) * Nn;
  float acc = 0.f;
  for (int n = threadIdx.x; n < Nn; n += 256) acc += row[n];
  __shared__ float red[256];
  red[threadIdx.x] = acc;
  __syncthreads();
  for (int st = 128; st > 0; st >>= 1) {
    if (threadIdx.x < st) red[threadIdx.x] += red[threadIdx.x + st];
    __syncthreads();
  }
  if (threadIdx.x == 0) sums[b * Sn + s] = red[0];
}

// ---------------- attn_bf[b][s][n] = bf16(attn/sums) ---------------------------
__global__ void k_attn_norm(const float* __restrict__ attn, const float* __restrict__ sums,
                            bf16_t* __restrict__ attn_bf) {
  const int b = blockIdx.z, s = blockIdx.y;
  const int n = blockIdx.x * 256 + threadIdx.x;
  const float inv = 1.f / sums[b * Sn + s];
  attn_bf[((size_t)b * 16 + s) * Nn + n] =
      (bf16_t)(attn[((size_t)b * Sn + s) * Nn + n] * inv);
}

// ---------------- updates[b][s][h] = sum_n attn[b,n,s] * vf[b,n,h] -------------
// A slab (16 x 4096 bf16) is shared by all 8 waves of the block -> TDM it into
// LDS in 16x512 chunks, double buffered, overlapping DMA with WMMA.
__global__ void __launch_bounds__(256) k_updates(const bf16_t* __restrict__ attn_bf,
                                                 const bf16_t* __restrict__ vf_t,
                                                 float* __restrict__ upd) {
  __shared__ __align__(16) bf16_t a_lds[2][16 * 512];  // 2 x 16KB double buffer
  const int b = blockIdx.y;
  const int wid = threadIdx.x >> 5;
  const int ht = blockIdx.x * 8 + wid;  // h-tile 0..15
  const bf16_t* Ag = attn_bf + (size_t)b * 16 * Nn;
  const bf16_t* Bb = vf_t + (size_t)b * Hn * Nn + (size_t)ht * 16 * Nn;

  if (wid == 0) {
    tdm_load_2d_bf16((unsigned int)(uintptr_t)&a_lds[0][0], Ag,
                     Nn, 16, 512, 16, Nn);
    __builtin_amdgcn_s_wait_tensorcnt(0);
  }
  __syncthreads();

  v8f acc = zero_v8f();
  const int nchunk = Nn / 512;  // 8
  for (int c = 0; c < nchunk; ++c) {
    if (wid == 0 && c + 1 < nchunk)  // prefetch next chunk via TDM while computing
      tdm_load_2d_bf16((unsigned int)(uintptr_t)&a_lds[(c + 1) & 1][0],
                       Ag + (c + 1) * 512, Nn, 16, 512, 16, Nn);
    const bf16_t* Ac = &a_lds[c & 1][0];
    const int c0 = c * 512;
#pragma unroll 4
    for (int k0 = 0; k0 < 512; k0 += 32) {
      __builtin_prefetch(Bb + c0 + k0 + 512, 0, 0);
      acc = wmma_bf16(load_frag_a_bf16(Ac + k0, 512),
                      load_frag_b_bf16(Bb + c0 + k0, Nn), acc);
    }
    if (wid == 0 && c + 1 < nchunk) __builtin_amdgcn_s_wait_tensorcnt(0);
    __syncthreads();
  }

  const int lane = threadIdx.x & 31, col = lane & 15, hf = lane >> 4;
#pragma unroll
  for (int i = 0; i < 8; ++i) {
    int s = i + hf * 8;
    if (s < Sn) upd[((size_t)b * Sn + s) * Hn + ht * 16 + col] = acc[i];
  }
}

// ---------------- GRU over s (sequential), single WG of 16 waves ---------------
// gx = updates@wi + b  (precomputed, [352][768]) ; wh_t bf16 [768][256]
__global__ void __launch_bounds__(512) k_gru(const float* __restrict__ gx,
                                             const bf16_t* __restrict__ wh_t,
                                             float* __restrict__ out) {
  __shared__ __align__(16) float  hbuf[32 * Hn];   // 32 KB  (carry, f32)
  __shared__ __align__(16) bf16_t zbuf[32 * Hn];   // 16 KB
  __shared__ __align__(16) bf16_t rhbuf[32 * Hn];  // 16 KB
  const int tid = threadIdx.x;
  for (int i = tid; i < 32 * Hn; i += 512) hbuf[i] = 0.f;
  __syncthreads();
  const int wid = tid >> 5, lane = tid & 31, col = lane & 15, hf = lane >> 4;

  for (int s = 0; s < Sn; ++s) {
    // ---- z / r phase: cols 0..511 of wh  (2 m-tiles x 32 n-tiles) ----
#pragma unroll 1
    for (int j = 0; j < 4; ++j) {
      const int t = wid + 16 * j;  // 0..63
      const int mt = t >> 5, nt = t & 31;
      v8f acc = zero_v8f();
      for (int k0 = 0; k0 < Hn; k0 += 32)
        acc = wmma_bf16(load_frag_a_f32(hbuf + mt * 16 * Hn + k0, Hn),
                        load_frag_b_bf16(wh_t + (size_t)nt * 16 * Hn + k0, Hn), acc);
#pragma unroll
      for (int i = 0; i < 8; ++i) {
        const int bb = mt * 16 + i + hf * 8;
        const int c = nt * 16 + col;
        const float val = acc[i] + gx[((size_t)bb * Sn + s) * 768 + c];
        const float sg = sigm(val);
        if (c < Hn) zbuf[bb * Hn + c] = (bf16_t)sg;
        else        rhbuf[bb * Hn + (c - Hn)] = (bf16_t)(sg * hbuf[bb * Hn + (c - Hn)]);
      }
    }
    __syncthreads();
    // ---- candidate phase: cols 512..767 (2 m-tiles x 16 n-tiles) ----
#pragma unroll 1
    for (int j = 0; j < 2; ++j) {
      const int t = wid + 16 * j;  // 0..31
      const int mt = t >> 4, nt = t & 15;
      v8f acc = zero_v8f();
      for (int k0 = 0; k0 < Hn; k0 += 32)
        acc = wmma_bf16(load_frag_a_bf16(rhbuf + mt * 16 * Hn + k0, Hn),
                        load_frag_b_bf16(wh_t + (size_t)(512 + nt * 16) * Hn + k0, Hn),
                        acc);
#pragma unroll
      for (int i = 0; i < 8; ++i) {
        const int bb = mt * 16 + i + hf * 8;
        const int c = nt * 16 + col;
        const float av = tanhf(acc[i] + gx[((size_t)bb * Sn + s) * 768 + 512 + c]);
        const float z  = (float)zbuf[bb * Hn + c];
        const float h0 = hbuf[bb * Hn + c];
        const float hn = h0 + z * (av - h0);
        hbuf[bb * Hn + c] = hn;
        out[((size_t)bb * Sn + s) * Hn + c] = hn;
      }
    }
    __syncthreads();
  }
}

// ================================ host side ====================================
extern "C" void kernel_launch(void* const* d_in, const int* in_sizes, int n_in,
                              void* d_out, int out_size, void* d_ws, size_t ws_size,
                              hipStream_t stream) {
  const float* inputs        = (const float*)d_in[0];
  const float* slot_noise    = (const float*)d_in[1];
  const float* ln_in_scale   = (const float*)d_in[2];
  const float* ln_in_offset  = (const float*)d_in[3];
  const float* ln_s_scale    = (const float*)d_in[4];
  const float* ln_s_offset   = (const float*)d_in[5];
  const float* ln_m_scale    = (const float*)d_in[6];
  const float* ln_m_offset   = (const float*)d_in[7];
  const float* slots_mu      = (const float*)d_in[8];
  const float* slots_ls      = (const float*)d_in[9];
  const float* Wq            = (const float*)d_in[10];
  const float* Wk            = (const float*)d_in[11];
  const float* Wv            = (const float*)d_in[12];
  const float* gru_wi        = (const float*)d_in[13];
  const float* gru_wh        = (const float*)d_in[14];
  const float* gru_b         = (const float*)d_in[15];
  const float* mlp_w1        = (const float*)d_in[16];
  const float* mlp_b1        = (const float*)d_in[17];
  const float* mlp_w2        = (const float*)d_in[18];
  const float* mlp_b2        = (const float*)d_in[19];

  char* ws = (char*)d_ws;
  auto alloc = [&](size_t bytes) -> void* {
    void* p = ws;
    ws += (bytes + 255) & ~(size_t)255;
    return p;
  };
  bf16_t* wk_t  = (bf16_t*)alloc((size_t)Hn * Dn * 2);
  bf16_t* wv_t  = (bf16_t*)alloc((size_t)Hn * Dn * 2);
  bf16_t* wq_t  = (bf16_t*)alloc((size_t)Hn * Hn * 2);
  bf16_t* wi_t  = (bf16_t*)alloc((size_t)768 * Hn * 2);
  bf16_t* wh_t  = (bf16_t*)alloc((size_t)768 * Hn * 2);
  bf16_t* w1_t  = (bf16_t*)alloc((size_t)Mn * Hn * 2);
  bf16_t* w2_t  = (bf16_t*)alloc((size_t)Hn * Mn * 2);
  bf16_t* kf    = (bf16_t*)alloc((size_t)Bn * Nn * Hn * 2);
  bf16_t* vf_t  = (bf16_t*)alloc((size_t)Bn * Hn * Nn * 2);
  bf16_t* qbuf  = (bf16_t*)alloc((size_t)Bn * 16 * Hn * 2);
  float*  logit = (float*) alloc((size_t)Bn * Sn * Nn * 4);
  bf16_t* attnb = (bf16_t*)alloc((size_t)Bn * 16 * Nn * 2);
  float*  sums  = (float*) alloc((size_t)Bn * Sn * 4);
  float*  slots = (float*) alloc((size_t)Bn * Sn * Hn * 4);
  float*  gruo  = (float*) alloc((size_t)Bn * Sn * Hn * 4);
  float*  lnb   = (float*) alloc((size_t)Bn * Sn * Hn * 4);
  float*  upd   = (float*) alloc((size_t)Bn * Sn * Hn * 4);
  float*  gxb   = (float*) alloc((size_t)Bn * Sn * 768 * 4);
  float*  hid   = (float*) alloc((size_t)Bn * Sn * Mn * 4);

  // one-time weight packing (transpose + bf16)
  k_transpose_bf16<<<(Dn * Hn + 255) / 256, 256, 0, stream>>>(Wk, wk_t, Dn, Hn);
  k_transpose_bf16<<<(Dn * Hn + 255) / 256, 256, 0, stream>>>(Wv, wv_t, Dn, Hn);
  k_transpose_bf16<<<(Hn * Hn + 255) / 256, 256, 0, stream>>>(Wq, wq_t, Hn, Hn);
  k_transpose_bf16<<<(Hn * 768 + 255) / 256, 256, 0, stream>>>(gru_wi, wi_t, Hn, 768);
  k_transpose_bf16<<<(Hn * 768 + 255) / 256, 256, 0, stream>>>(gru_wh, wh_t, Hn, 768);
  k_transpose_bf16<<<(Hn * Mn + 255) / 256, 256, 0, stream>>>(mlp_w1, w1_t, Hn, Mn);
  k_transpose_bf16<<<(Mn * Hn + 255) / 256, 256, 0, stream>>>(mlp_w2, w2_t, Mn, Hn);

  k_init_slots<<<(Bn * Sn * Hn) / 256, 256, 0, stream>>>(slot_noise, slots_mu, slots_ls,
                                                         slots);

  // fused input LN + kf/vf projections (the big GEMMs)
  k_ln_project<<<dim3(Nn / 16, Bn), 256, 0, stream>>>(inputs, ln_in_scale, ln_in_offset,
                                                      wk_t, wv_t, kf, vf_t);

  const int Mr = Bn * Sn;  // 352 rows (= 22 tiles of 16)
  for (int it = 0; it < 3; ++it) {
    k_ln_rows<<<Mr, 256, 0, stream>>>(slots, ln_s_scale, ln_s_offset, lnb);
    k_qproj<<<(22 * 16) / 8, 256, 0, stream>>>(lnb, wq_t, qbuf);
    k_logits<<<dim3(Nn / 16 / 8, Bn), 256, 0, stream>>>(kf, qbuf, logit);
    k_softmax_s<<<dim3(Nn / 256, Bn), 256, 0, stream>>>(logit);
    k_rowsum<<<dim3(Sn, Bn), 256, 0, stream>>>(logit, sums);
    k_attn_norm<<<dim3(Nn / 256, Sn, Bn), 256, 0, stream>>>(logit, sums, attnb);
    k_updates<<<dim3(2, Bn), 256, 0, stream>>>(attnb, vf_t, upd);
    k_gemm<<<(22 * 48) / 8, 256, 0, stream>>>(upd, wi_t, gru_b, nullptr, gxb, Mr, Hn,
                                              768, 0);
    k_gru<<<1, 512, 0, stream>>>(gxb, wh_t, gruo);
    k_ln_rows<<<Mr, 256, 0, stream>>>(gruo, ln_m_scale, ln_m_offset, lnb);
    k_gemm<<<(22 * 32) / 8, 256, 0, stream>>>(lnb, w1_t, mlp_b1, nullptr, hid, Mr, Hn,
                                              Mn, 1);
    float* dest = (it == 2) ? (float*)d_out : slots;
    k_gemm<<<(22 * 16) / 8, 256, 0, stream>>>(hid, w2_t, mlp_b2, gruo, dest, Mr, Mn, Hn,
                                              0);
  }
}
